// MultiHeadAttention_36051955482775
// MI455X (gfx1250) — compile-verified, matrix operand order checked by reading, not run on hardware
//
#include <hip/hip_runtime.h>

// Shapes fixed by the reference.
#define B_   4
#define S_   2048
#define D_   1024
#define H_   16
#define DK_  64
#define M_   (B_ * S_)   // 8192
#define K_   1024

typedef __attribute__((ext_vector_type(16))) _Float16 v16h;
typedef __attribute__((ext_vector_type(8)))  _Float16 v8h;
typedef __attribute__((ext_vector_type(8)))  float    v8f;
typedef __attribute__((ext_vector_type(4)))  float    v4f;

__device__ __forceinline__ v8f wmma16x16x32(v16h a, v16h b, v8f c) {
  // D = A(16x32 f16) * B(32x16 f16) + C(16x16 f32)
  return __builtin_amdgcn_wmma_f32_16x16x32_f16(false, a, false, b, (short)0, c,
                                                false, false);
}

// Build a 16x32 f16 A/B fragment from LDS per the CDNA5 VGPR layout:
// lane hl holds row/col hl; halves 0..7 = K {kbase + hi*8 ..}, halves 8..15 = +16.
__device__ __forceinline__ v16h frag_lds(const _Float16* base, int row, int stride,
                                         int kbase, int hi) {
  const _Float16* p = base + row * stride + kbase + hi * 8;
  v8h lo = *(const v8h*)(p);
  v8h hv = *(const v8h*)(p + 16);
  v16h f;
#pragma unroll
  for (int i = 0; i < 8; ++i) { f[i] = lo[i]; f[i + 8] = hv[i]; }
  return f;
}

// Wave-relative LDS byte address: flat shared addresses map to LDS via addr[31:0].
__device__ __forceinline__ unsigned lds_addr32(const void* p) {
  return (unsigned)(unsigned long long)p;
}

// Async 16B global -> LDS copy, tracked by ASYNCcnt (CDNA5 TDM-family path).
__device__ __forceinline__ void async_copy_b128(unsigned lds_dst, const void* gsrc) {
  asm volatile("global_load_async_to_lds_b128 %0, %1, off"
               :: "v"(lds_dst), "v"(gsrc)
               : "memory");
}
__device__ __forceinline__ void wait_asynccnt0() {
  asm volatile("s_wait_asynccnt 0x0" ::: "memory");
}

// C = X[M,K] * W[N,K]^T + bias.  Block tile 128x64, 8 waves (4x2), wave tile 32x32.
// OUT_MODE 0: f32 row-major [M, D_].  OUT_MODE 1: f16 head-split [B,H,S,DK].
template <int OUT_MODE>
__global__ __launch_bounds__(256) void gemm_bias_wmma(const float* __restrict__ X,
                                                      const float* __restrict__ W,
                                                      const float* __restrict__ bias,
                                                      void* __restrict__ outp) {
  constexpr int LDA = 40, LDB = 40;  // +8-half pad: 80B row stride, conflict-friendly
  __shared__ __align__(16) _Float16 ldsA[128 * LDA];
  __shared__ __align__(16) _Float16 ldsB[64 * LDB];
  const int tid = threadIdx.x;
  const int wave = tid >> 5, lane = tid & 31;
  const int hl = lane & 15, hi = lane >> 4;
  const int wm = wave >> 1, wn = wave & 1;
  const int bm = blockIdx.x * 128, bn = blockIdx.y * 64;

  const int xr = tid >> 1, xc = (tid & 1) * 16;  // X stage: 128 rows x 32 cols
  const int wr = tid >> 2, wc = (tid & 3) * 8;   // W stage: 64 rows x 32 cols

  v8f acc[2][2] = {};

  for (int k0 = 0; k0 < K_; k0 += 32) {
    // Stage X tile, f32 -> f16 once (conversion forces the VGPR path here).
    const v4f* xg = (const v4f*)(X + (size_t)(bm + xr) * K_ + k0 + xc);
    v4f x0 = xg[0], x1 = xg[1], x2 = xg[2], x3 = xg[3];
    v8h hx0, hx1;
#pragma unroll
    for (int i = 0; i < 4; ++i) {
      hx0[i] = (_Float16)x0[i]; hx0[i + 4] = (_Float16)x1[i];
      hx1[i] = (_Float16)x2[i]; hx1[i + 4] = (_Float16)x3[i];
    }
    *(v8h*)(ldsA + xr * LDA + xc)     = hx0;
    *(v8h*)(ldsA + xr * LDA + xc + 8) = hx1;
    // Stage W tile.
    const v4f* wg = (const v4f*)(W + (size_t)(bn + wr) * K_ + k0 + wc);
    v4f w0 = wg[0], w1 = wg[1];
    v8h hw;
#pragma unroll
    for (int i = 0; i < 4; ++i) { hw[i] = (_Float16)w0[i]; hw[i + 4] = (_Float16)w1[i]; }
    *(v8h*)(ldsB + wr * LDB + wc) = hw;

    if (k0 + 32 < K_) {  // hint next K-slice into cache (global_prefetch_b8)
      __builtin_prefetch(X + (size_t)(bm + xr) * K_ + k0 + 32 + xc, 0, 1);
      __builtin_prefetch(W + (size_t)(bn + wr) * K_ + k0 + 32 + wc, 0, 1);
    }
    __syncthreads();

    v16h a0 = frag_lds(ldsA, wm * 32 + hl,      LDA, 0, hi);
    v16h a1 = frag_lds(ldsA, wm * 32 + 16 + hl, LDA, 0, hi);
    v16h b0 = frag_lds(ldsB, wn * 32 + hl,      LDB, 0, hi);
    v16h b1 = frag_lds(ldsB, wn * 32 + 16 + hl, LDB, 0, hi);
    acc[0][0] = wmma16x16x32(a0, b0, acc[0][0]);
    acc[0][1] = wmma16x16x32(a0, b1, acc[0][1]);
    acc[1][0] = wmma16x16x32(a1, b0, acc[1][0]);
    acc[1][1] = wmma16x16x32(a1, b1, acc[1][1]);
    __syncthreads();
  }

  float bvv[2];
#pragma unroll
  for (int ni = 0; ni < 2; ++ni) bvv[ni] = bias[bn + wn * 32 + ni * 16 + hl];

#pragma unroll
  for (int mi = 0; mi < 2; ++mi)
#pragma unroll
    for (int ni = 0; ni < 2; ++ni)
#pragma unroll
      for (int r = 0; r < 8; ++r) {
        float c = acc[mi][ni][r] + bvv[ni];
        int m = bm + wm * 32 + mi * 16 + 8 * hi + r;  // C layout: vgpr r, half-wave hi
        int n = bn + wn * 32 + ni * 16 + hl;
        if (OUT_MODE == 0) {
          ((float*)outp)[(size_t)m * D_ + n] = c;
        } else {
          int bb = m >> 11, ss = m & (S_ - 1);
          int hh = n >> 6,  dk = n & (DK_ - 1);
          ((_Float16*)outp)[((((size_t)bb * H_) + hh) * S_ + ss) * DK_ + dk] =
              (_Float16)c;
        }
      }
}

// Flash attention: one block = one (b,h) x 128 q rows; 8 waves x 16 q rows each.
// K tiles stream via async global->LDS (ASYNCcnt); V is register-transposed.
__global__ __launch_bounds__(256) void flash_attn_wmma(
    const _Float16* __restrict__ qh, const _Float16* __restrict__ kh,
    const _Float16* __restrict__ vh, float* __restrict__ combined,
    float* __restrict__ mOut, float* __restrict__ lOut) {
  constexpr int LDK = 72;  // 64 + 8 halves pad (144B rows)
  __shared__ __align__(16) _Float16 ldsK[64 * LDK];
  __shared__ __align__(16) _Float16 ldsVT[64 * LDK];      // V transposed: [dk][kv]
  __shared__ __align__(16) _Float16 ldsP[8 * 16 * LDK];   // per-wave P re-layout slab
  const int tid = threadIdx.x;
  const int wave = tid >> 5, lane = tid & 31;
  const int hl = lane & 15, hi = lane >> 4;
  const int bh = blockIdx.y;
  const int b = bh / H_, h = bh % H_;
  const int qt = blockIdx.x;

  // Load the wave's q A-fragments once; fold softmax scale 1/sqrt(DK).
  const int qrow = qt * 128 + wave * 16 + hl;
  const _Float16* qp = qh + ((size_t)bh * S_ + qrow) * DK_;
  v16h aq[2];
#pragma unroll
  for (int ks = 0; ks < 2; ++ks) {
    const _Float16* p = qp + ks * 32 + hi * 8;
    v8h lo = *(const v8h*)p;
    v8h hv = *(const v8h*)(p + 16);
#pragma unroll
    for (int i = 0; i < 8; ++i) {
      aq[ks][i]     = lo[i] * (_Float16)0.125f;
      aq[ks][i + 8] = hv[i] * (_Float16)0.125f;
    }
  }

  float mrow[8], lrow[8];
  v8f o[4] = {};
#pragma unroll
  for (int r = 0; r < 8; ++r) { mrow[r] = -3.0e38f; lrow[r] = 0.f; }

  const int sr = tid >> 2, sc = (tid & 3) * 16;  // KV staging map
  _Float16* pw = ldsP + wave * 16 * LDK;
  const unsigned kDst0 = lds_addr32(ldsK + sr * LDK + sc);
  const unsigned kDst1 = lds_addr32(ldsK + sr * LDK + sc + 8);

  for (int t = 0; t < 32; ++t) {
    // K tile: raw f16 bytes, async global -> LDS (no VGPR round-trip).
    const _Float16* kgp = kh + ((size_t)bh * S_ + t * 64 + sr) * DK_ + sc;
    async_copy_b128(kDst0, kgp);
    async_copy_b128(kDst1, kgp + 8);
    // V tile: needs a transpose, so stage through VGPRs.
    const v8h* vg = (const v8h*)(vh + ((size_t)bh * S_ + t * 64 + sr) * DK_ + sc);
    v8h vv0 = vg[0], vv1 = vg[1];
#pragma unroll
    for (int i = 0; i < 8; ++i) {
      ldsVT[(sc + i)     * LDK + sr] = vv0[i];
      ldsVT[(sc + 8 + i) * LDK + sr] = vv1[i];
    }
    wait_asynccnt0();
    __syncthreads();

    // Scores S[16 x 64]: 4 column tiles x 2 K-steps over DK=64.
    v8f s[4];
#pragma unroll
    for (int j = 0; j < 4; ++j) {
      v8f a = {};
#pragma unroll
      for (int ks = 0; ks < 2; ++ks) {
        v16h bk = frag_lds(ldsK, j * 16 + hl, LDK, ks * 32, hi);
        a = wmma16x16x32(aq[ks], bk, a);
      }
      s[j] = a;
    }

    // Online softmax: row max / sum via width-16 xor-shuffles (rows split per half-wave).
    float alpha[8];
#pragma unroll
    for (int r = 0; r < 8; ++r) {
      float mx = fmaxf(fmaxf(s[0][r], s[1][r]), fmaxf(s[2][r], s[3][r]));
#pragma unroll
      for (int off = 1; off < 16; off <<= 1) mx = fmaxf(mx, __shfl_xor(mx, off, 16));
      float mnew = fmaxf(mrow[r], mx);
      alpha[r] = __expf(mrow[r] - mnew);
      mrow[r] = mnew;
    }
#pragma unroll
    for (int j = 0; j < 4; ++j)
#pragma unroll
      for (int r = 0; r < 8; ++r) s[j][r] = __expf(s[j][r] - mrow[r]);
#pragma unroll
    for (int r = 0; r < 8; ++r) {
      float rs = s[0][r] + s[1][r] + s[2][r] + s[3][r];
#pragma unroll
      for (int off = 1; off < 16; off <<= 1) rs += __shfl_xor(rs, off, 16);
      lrow[r] = lrow[r] * alpha[r] + rs;
    }
#pragma unroll
    for (int j = 0; j < 4; ++j)
#pragma unroll
      for (int r = 0; r < 8; ++r) o[j][r] *= alpha[r];

    // Re-layout P (C-frag) -> A-frag via per-wave LDS; same-wave LDS is in-order,
    // just drain DScnt before the fragment reads.
#pragma unroll
    for (int j = 0; j < 4; ++j)
#pragma unroll
      for (int r = 0; r < 8; ++r)
        pw[(r + 8 * hi) * LDK + j * 16 + hl] = (_Float16)s[j][r];
    asm volatile("s_wait_dscnt 0x0" ::: "memory");

    v16h ap0 = frag_lds(pw, hl, LDK, 0, hi);
    v16h ap1 = frag_lds(pw, hl, LDK, 32, hi);
#pragma unroll
    for (int j = 0; j < 4; ++j) {
      v16h bv0 = frag_lds(ldsVT, j * 16 + hl, LDK, 0, hi);
      o[j] = wmma16x16x32(ap0, bv0, o[j]);
      v16h bv1 = frag_lds(ldsVT, j * 16 + hl, LDK, 32, hi);
      o[j] = wmma16x16x32(ap1, bv1, o[j]);
    }
    __syncthreads();
  }

  float inv[8];
#pragma unroll
  for (int r = 0; r < 8; ++r) inv[r] = 1.0f / lrow[r];
#pragma unroll
  for (int j = 0; j < 4; ++j)
#pragma unroll
    for (int r = 0; r < 8; ++r) {
      int qg = qt * 128 + wave * 16 + 8 * hi + r;
      combined[((size_t)b * S_ + qg) * D_ + h * DK_ + j * 16 + hl] = o[j][r] * inv[r];
    }
  if (h == H_ - 1 && hl == 0) {
#pragma unroll
    for (int r = 0; r < 8; ++r) {
      int qg = qt * 128 + wave * 16 + 8 * hi + r;
      mOut[b * S_ + qg] = mrow[r];
      lOut[b * S_ + qg] = lrow[r];
    }
  }
}

// Second pass for the last head only: recompute scores with WMMA and emit
// normalized probabilities using the saved (m, l).
__global__ __launch_bounds__(256) void attn_last_wmma(
    const _Float16* __restrict__ qh, const _Float16* __restrict__ kh,
    const float* __restrict__ mIn, const float* __restrict__ lIn,
    float* __restrict__ attn) {
  constexpr int LDK = 72;
  __shared__ __align__(16) _Float16 ldsK[64 * LDK];
  const int tid = threadIdx.x;
  const int wave = tid >> 5, lane = tid & 31;
  const int hl = lane & 15, hi = lane >> 4;
  const int b = blockIdx.y;
  const int bh = b * H_ + (H_ - 1);
  const int qt = blockIdx.x;

  const int qrow = qt * 128 + wave * 16 + hl;
  const _Float16* qp = qh + ((size_t)bh * S_ + qrow) * DK_;
  v16h aq[2];
#pragma unroll
  for (int ks = 0; ks < 2; ++ks) {
    const _Float16* p = qp + ks * 32 + hi * 8;
    v8h lo = *(const v8h*)p;
    v8h hv = *(const v8h*)(p + 16);
#pragma unroll
    for (int i = 0; i < 8; ++i) {
      aq[ks][i]     = lo[i] * (_Float16)0.125f;
      aq[ks][i + 8] = hv[i] * (_Float16)0.125f;
    }
  }

  float mrow[8], ilrow[8];
#pragma unroll
  for (int r = 0; r < 8; ++r) {
    int qg = qt * 128 + wave * 16 + 8 * hi + r;
    mrow[r]  = mIn[b * S_ + qg];
    ilrow[r] = 1.0f / lIn[b * S_ + qg];
  }

  const int sr = tid >> 2, sc = (tid & 3) * 16;
  const unsigned kDst0 = lds_addr32(ldsK + sr * LDK + sc);
  const unsigned kDst1 = lds_addr32(ldsK + sr * LDK + sc + 8);
  for (int t = 0; t < 32; ++t) {
    const _Float16* kgp = kh + ((size_t)bh * S_ + t * 64 + sr) * DK_ + sc;
    async_copy_b128(kDst0, kgp);
    async_copy_b128(kDst1, kgp + 8);
    wait_asynccnt0();
    __syncthreads();
#pragma unroll
    for (int j = 0; j < 4; ++j) {
      v8f a = {};
#pragma unroll
      for (int ks = 0; ks < 2; ++ks) {
        v16h bk = frag_lds(ldsK, j * 16 + hl, LDK, ks * 32, hi);
        a = wmma16x16x32(aq[ks], bk, a);
      }
#pragma unroll
      for (int r = 0; r < 8; ++r) {
        int qg = qt * 128 + wave * 16 + 8 * hi + r;
        attn[((size_t)b * S_ + qg) * S_ + t * 64 + j * 16 + hl] =
            __expf(a[r] - mrow[r]) * ilrow[r];
      }
    }
    __syncthreads();
  }
}

extern "C" void kernel_launch(void* const* d_in, const int* in_sizes, int n_in,
                              void* d_out, int out_size, void* d_ws, size_t ws_size,
                              hipStream_t stream) {
  (void)in_sizes; (void)n_in; (void)out_size; (void)ws_size;
  const float* Q  = (const float*)d_in[0];
  const float* K  = (const float*)d_in[1];
  const float* V  = (const float*)d_in[2];
  const float* Wq = (const float*)d_in[3]; const float* bq = (const float*)d_in[4];
  const float* Wk = (const float*)d_in[5]; const float* bk = (const float*)d_in[6];
  const float* Wv = (const float*)d_in[7]; const float* bv = (const float*)d_in[8];
  const float* Wo = (const float*)d_in[9]; const float* bo = (const float*)d_in[10];
  float* out = (float*)d_out;

  // Workspace: f16 q/k/v in [B,H,S,DK] (16 MB each) + last-head (m,l).
  const size_t nBHSD = (size_t)B_ * H_ * S_ * DK_;  // 8Mi elements
  _Float16* qhp = (_Float16*)d_ws;
  _Float16* khp = qhp + nBHSD;
  _Float16* vhp = khp + nBHSD;
  float* mL = (float*)(vhp + nBHSD);
  float* lL = mL + (size_t)B_ * S_;

  // Scratch the f32 "combined" [B,S,D] inside d_out's attn region; it is fully
  // consumed by the final GEMM, then fully overwritten by attn_last_wmma.
  float* attnOut  = out + (size_t)B_ * S_ * D_;
  float* combined = attnOut;

  dim3 gg(M_ / 128, D_ / 64), blk(256);
  gemm_bias_wmma<1><<<gg, blk, 0, stream>>>(Q, Wq, bq, (void*)qhp);
  gemm_bias_wmma<1><<<gg, blk, 0, stream>>>(K, Wk, bk, (void*)khp);
  gemm_bias_wmma<1><<<gg, blk, 0, stream>>>(V, Wv, bv, (void*)vhp);
  flash_attn_wmma<<<dim3(S_ / 128, B_ * H_), blk, 0, stream>>>(qhp, khp, vhp,
                                                               combined, mL, lL);
  gemm_bias_wmma<0><<<gg, blk, 0, stream>>>(combined, Wo, bo, (void*)out);
  attn_last_wmma<<<dim3(S_ / 128, B_), blk, 0, stream>>>(qhp, khp, mL, lL, attnOut);
}
